// AutoEPMoELayer_89842125898081
// MI455X (gfx1250) — matrix operand order repaired and checked by reading, not compiled, not run on hardware
//
#include <hip/hip_runtime.h>

// MoE SwiGLU for gfx1250 (MI455X): router -> sort-by-expert -> grouped GEMM
// (bf16 WMMA, fp32 accum) -> atomic scatter-combine.
// Weights stream 192 MB fp32 once from HBM (fits 192 MB global L2; per-expert
// re-reads across token tiles hit L2). Math in v_wmma_f32_16x16x32_bf16;
// staging converts via v_cvt_pk_bf16_f32 + v_perm_b32; LDS double-buffered.

#define E_EXPERTS 8
#define TOPK      2
#define DIM       1024
#define FFN       2048
#define NTOK      1024
#define CAP       2560   // 2048 slots + per-expert align-64 padding
#define TM        64     // token tile (4 x 16-row wave groups)
#define TN        64     // column tile (2 wave col-groups x 2 frags x 16)
#define KSTEP     32     // bf16 WMMA K
#define KPAD      40     // LDS col stride (elements): 16B aligned, low-conflict
#define MAXT      32     // max 64-token tiles per expert (2048/64)

typedef __attribute__((ext_vector_type(16))) __bf16 v16bf;
typedef __attribute__((ext_vector_type(2)))  __bf16 v2bf;
typedef __attribute__((ext_vector_type(8)))  float  v8f;

union BF16Frag { v16bf v; uint4 q[2]; };

__device__ __forceinline__ unsigned short f2bf(float f) {
  union { __bf16 b; unsigned short u; } cv;
  cv.b = (__bf16)f;                    // native bf16 convert on gfx1250
  return cv.u;
}

// adjacent pack {bf(a)[15:0], bf(b)[31:16]} -> single v_cvt_pk_bf16_f32
__device__ __forceinline__ unsigned int cvtpk(float a, float b) {
  v2bf v = {(__bf16)a, (__bf16)b};
  return __builtin_bit_cast(unsigned int, v);
}

__device__ __forceinline__ v8f vzero() {
  v8f z = {0.f, 0.f, 0.f, 0.f, 0.f, 0.f, 0.f, 0.f};
  return z;
}

__device__ __forceinline__ v8f wmma_bf16(v16bf a, v16bf b, v8f c) {
  return __builtin_amdgcn_wmma_f32_16x16x32_bf16(false, a, false, b,
                                                 (short)0, c, false, false);
}

// A fragment (16x32 bf16, row-major source, stride ld elements).
// ISA layout: lane L -> row M = L&15; lane<16 holds K{0..7,16..23},
// lane>=16 holds K{8..15,24..31}. Two 16B loads per lane.
__device__ __forceinline__ v16bf load_frag_a(const unsigned short* p, int ld, int lane) {
  int m  = lane & 15;
  int kh = (lane < 16) ? 0 : 8;
  const unsigned short* q = p + (size_t)m * ld + kh;
  BF16Frag f;
  f.q[0] = *reinterpret_cast<const uint4*>(q);        // K kh..kh+7
  f.q[1] = *reinterpret_cast<const uint4*>(q + 16);   // K kh+16..kh+23
  return f.v;
}

// B fragment (32x16 bf16) from LDS tile staged column-major [col][KPAD].
// ISA layout: lane L -> col N = L&15; lane<16 holds K{0..7,16..23}, lane>=16 K{8..15,24..31}.
__device__ __forceinline__ v16bf load_frag_b(const unsigned short* base, int lane) {
  int n  = lane & 15;
  int kh = (lane < 16) ? 0 : 8;
  const unsigned short* q = base + n * KPAD + kh;
  BF16Frag f;
  f.q[0] = *reinterpret_cast<const uint4*>(q);
  f.q[1] = *reinterpret_cast<const uint4*>(q + 16);
  return f.v;
}

// pack two k-rows (lo=k, hi=k+1) of 4 columns into the [col][k] LDS tile.
// 2 VALU/word: natural-adjacency v_cvt_pk_bf16_f32 then v_perm_b32 cross-select.
__device__ __forceinline__ void stage_pair(unsigned short* dst, int rp, int sc,
                                           float4 lo, float4 hi) {
  unsigned int la = cvtpk(lo.x, lo.y);   // {bf(lo.x), bf(lo.y)}
  unsigned int lb = cvtpk(lo.z, lo.w);
  unsigned int ha = cvtpk(hi.x, hi.y);   // {bf(hi.x), bf(hi.y)}
  unsigned int hb = cvtpk(hi.z, hi.w);
  // perm(a,b,sel): sel codes 0-3 pick bytes of b, 4-7 pick bytes of a
  unsigned int w0 = __builtin_amdgcn_perm(ha, la, 0x05040100u); // {bf(lo.x),bf(hi.x)}
  unsigned int w1 = __builtin_amdgcn_perm(ha, la, 0x07060302u); // {bf(lo.y),bf(hi.y)}
  unsigned int w2 = __builtin_amdgcn_perm(hb, lb, 0x05040100u); // {bf(lo.z),bf(hi.z)}
  unsigned int w3 = __builtin_amdgcn_perm(hb, lb, 0x07060302u); // {bf(lo.w),bf(hi.w)}
  unsigned int* p = reinterpret_cast<unsigned int*>(dst);
  p[((sc + 0) * KPAD >> 1) + rp] = w0;
  p[((sc + 1) * KPAD >> 1) + rp] = w1;
  p[((sc + 2) * KPAD >> 1) + rp] = w2;
  p[((sc + 3) * KPAD >> 1) + rp] = w3;
}

__global__ void zero_u32(unsigned int* __restrict__ p, long n) {
  long i = (long)blockIdx.x * blockDim.x + threadIdx.x;
  long stride = (long)gridDim.x * blockDim.x;
  for (; i < n; i += stride) p[i] = 0u;
}

// ---------------- router: softmax over 8 experts, top-2 ----------------
__global__ __launch_bounds__(256) void moe_router(
    const float* __restrict__ x, const float* __restrict__ gate_w,
    int* __restrict__ tok_sel, float* __restrict__ tok_score,
    int* __restrict__ counts) {
  int wave = threadIdx.x >> 5;
  int lane = threadIdx.x & 31;
  int token = blockIdx.x * 8 + wave;
  float acc[E_EXPERTS];
#pragma unroll
  for (int e = 0; e < E_EXPERTS; ++e) acc[e] = 0.f;
  const float* xr = x + (size_t)token * DIM;
#pragma unroll
  for (int j = 0; j < DIM / 128; ++j) {
    int idx = j * 128 + lane * 4;
    float4 xv = *reinterpret_cast<const float4*>(xr + idx);
#pragma unroll
    for (int e = 0; e < E_EXPERTS; ++e) {
      float4 gv = *reinterpret_cast<const float4*>(gate_w + e * DIM + idx);
      acc[e] += xv.x * gv.x + xv.y * gv.y + xv.z * gv.z + xv.w * gv.w;
    }
  }
#pragma unroll
  for (int off = 16; off > 0; off >>= 1)
#pragma unroll
    for (int e = 0; e < E_EXPERTS; ++e) acc[e] += __shfl_xor(acc[e], off, 32);
  if (lane == 0) {
    float m = acc[0];
#pragma unroll
    for (int e = 1; e < E_EXPERTS; ++e) m = fmaxf(m, acc[e]);
    float p[E_EXPERTS];
    float s = 0.f;
#pragma unroll
    for (int e = 0; e < E_EXPERTS; ++e) { p[e] = __expf(acc[e] - m); s += p[e]; }
    float inv = 1.f / s;
    int i0 = 0; float b0 = -1.f;
#pragma unroll
    for (int e = 0; e < E_EXPERTS; ++e) {
      float sc = p[e] * inv;
      if (sc > b0) { b0 = sc; i0 = e; }
    }
    int i1 = (i0 == 0) ? 1 : 0; float b1 = -1.f;
#pragma unroll
    for (int e = 0; e < E_EXPERTS; ++e) {
      if (e == i0) continue;
      float sc = p[e] * inv;
      if (sc > b1) { b1 = sc; i1 = e; }
    }
    tok_sel[token * 2 + 0] = i0;
    tok_sel[token * 2 + 1] = i1;
    tok_score[token * 2 + 0] = b0;
    tok_score[token * 2 + 1] = b1;
    atomicAdd(&counts[i0], 1);
    atomicAdd(&counts[i1], 1);
  }
}

// exclusive scan of counts, offsets aligned to TM so tiles never cross experts
__global__ void moe_offsets(const int* __restrict__ counts, int* __restrict__ offsets) {
  if (threadIdx.x == 0 && blockIdx.x == 0) {
    int off = 0;
    for (int e = 0; e < E_EXPERTS; ++e) {
      offsets[e] = off;
      off += (counts[e] + (TM - 1)) & ~(TM - 1);
    }
  }
}

// ------------- assign slots + gather x rows (fp32 -> bf16) -------------
__global__ __launch_bounds__(256) void moe_assign(
    const float* __restrict__ x,
    const int* __restrict__ tok_sel, const float* __restrict__ tok_score,
    const int* __restrict__ offsets, int* __restrict__ cursors,
    int* __restrict__ slot_token, float* __restrict__ slot_score,
    unsigned short* __restrict__ x_sorted) {
  int wave = threadIdx.x >> 5;
  int lane = threadIdx.x & 31;
  int a = blockIdx.x * 8 + wave;        // routed-slot assignment id, 0..2047
  int token = a & (NTOK - 1);
  int kk = a >> 10;
  int slot = 0;
  if (lane == 0) {
    int e = tok_sel[token * 2 + kk];
    int pos = atomicAdd(&cursors[e], 1);
    slot = offsets[e] + pos;
    slot_token[slot] = token;
    slot_score[slot] = tok_score[token * 2 + kk];
  }
  slot = __shfl(slot, 0, 32);
  const float* xr = x + (size_t)token * DIM;
  unsigned short* dst = x_sorted + (size_t)slot * DIM;
#pragma unroll
  for (int j = 0; j < DIM / 128; ++j) {
    int idx = j * 128 + lane * 4;
    float4 v = *reinterpret_cast<const float4*>(xr + idx);
    uint2 o;
    o.x = cvtpk(v.x, v.y);              // adjacency-natural -> v_cvt_pk_bf16_f32
    o.y = cvtpk(v.z, v.w);
    *reinterpret_cast<uint2*>(dst + idx) = o;
  }
}

// ---------- phase 1: h = silu(x_e @ w1[e]) * (x_e @ w3[e]) -> bf16 ----------
__global__ __launch_bounds__(256) void moe_phase1(
    const unsigned short* __restrict__ x_sorted,
    const float* __restrict__ w1, const float* __restrict__ w3,
    unsigned short* __restrict__ hbuf,
    const int* __restrict__ counts, const int* __restrict__ offsets) {
  int e = blockIdx.y >> 5;
  int t = blockIdx.y & (MAXT - 1);
  int cnt = counts[e];
  if (t * TM >= cnt) return;            // uniform: whole block exits together
  int row0 = offsets[e] + t * TM;
  int colblk = blockIdx.x * TN;

  __shared__ unsigned short Bs1[2][TN * KPAD];
  __shared__ unsigned short Bs3[2][TN * KPAD];

  int tid = threadIdx.x;
  int wave = tid >> 5, lane = tid & 31;
  int wrow = (wave >> 1) * 16;          // 0,16,32,48 within token tile
  int wcol = (wave & 1) * 32;           // 0,32 within column tile

  const float* w1e = w1 + (size_t)e * DIM * FFN + colblk;
  const float* w3e = w3 + (size_t)e * DIM * FFN + colblk;
  const unsigned short* arow = x_sorted + (size_t)(row0 + wrow) * DIM;

  // staging map: thread -> k rows (2rp, 2rp+1) x 4 cols
  int rp = tid >> 4;                    // 0..15
  int sc = (tid & 15) * 4;              // col 0..60

  v8f acc1[2] = {vzero(), vzero()};
  v8f acc3[2] = {vzero(), vzero()};

  // prologue: stage tile k0=0 into buffer 0
  {
    size_t g0 = (size_t)(2 * rp) * FFN + sc;
    stage_pair(Bs1[0], rp, sc, *reinterpret_cast<const float4*>(w1e + g0),
               *reinterpret_cast<const float4*>(w1e + g0 + FFN));
    stage_pair(Bs3[0], rp, sc, *reinterpret_cast<const float4*>(w3e + g0),
               *reinterpret_cast<const float4*>(w3e + g0 + FFN));
  }
  __syncthreads();

  for (int k0 = 0; k0 < DIM; k0 += KSTEP) {
    int cur = (k0 >> 5) & 1;
    bool has_next = (k0 + KSTEP) < DIM;
    float4 n1a, n1b, n3a, n3b;
    if (has_next) {                     // issue next tile's global loads early
      size_t g0 = (size_t)(k0 + KSTEP + 2 * rp) * FFN + sc;
      n1a = *reinterpret_cast<const float4*>(w1e + g0);
      n1b = *reinterpret_cast<const float4*>(w1e + g0 + FFN);
      n3a = *reinterpret_cast<const float4*>(w3e + g0);
      n3b = *reinterpret_cast<const float4*>(w3e + g0 + FFN);
      if (k0 + 2 * KSTEP < DIM)         // and hint the tile after toward L2
        __builtin_prefetch(w1e + (size_t)(k0 + 2 * KSTEP) * FFN, 0, 1);
    }
    // compute on current buffer: all fragments live -> overlapped ds loads
    v16bf a   = load_frag_a(arow + k0, DIM, lane);
    v16bf b1a = load_frag_b(&Bs1[cur][(wcol + 0) * KPAD], lane);
    v16bf b3a = load_frag_b(&Bs3[cur][(wcol + 0) * KPAD], lane);
    v16bf b1b = load_frag_b(&Bs1[cur][(wcol + 16) * KPAD], lane);
    v16bf b3b = load_frag_b(&Bs3[cur][(wcol + 16) * KPAD], lane);
    acc1[0] = wmma_bf16(a, b1a, acc1[0]);
    acc3[0] = wmma_bf16(a, b3a, acc3[0]);
    acc1[1] = wmma_bf16(a, b1b, acc1[1]);
    acc3[1] = wmma_bf16(a, b3b, acc3[1]);
    if (has_next) {                     // convert + store into other buffer
      stage_pair(Bs1[cur ^ 1], rp, sc, n1a, n1b);
      stage_pair(Bs3[cur ^ 1], rp, sc, n3a, n3b);
    }
    __syncthreads();                    // one barrier per k-step
  }
  // fused SwiGLU epilogue; C/D layout: lane holds col N=lane&15, rows j+(lane<16?0:8)
  int rbase = row0 + wrow + ((lane < 16) ? 0 : 8);
#pragma unroll
  for (int f = 0; f < 2; ++f) {
    int col = colblk + wcol + f * 16 + (lane & 15);
#pragma unroll
    for (int r = 0; r < 8; ++r) {
      float g = acc1[f][r];
      float sg = g * __builtin_amdgcn_rcpf(1.0f + __expf(-g));  // fast silu
      hbuf[(size_t)(rbase + r) * FFN + col] = f2bf(sg * acc3[f][r]);
    }
  }
}

// --------- phase 2: y = h @ w2[e]; out[token] += y * score (atomic) ---------
__global__ __launch_bounds__(256) void moe_phase2(
    const unsigned short* __restrict__ hbuf,
    const float* __restrict__ w2,
    const int* __restrict__ slot_token, const float* __restrict__ slot_score,
    float* __restrict__ out,
    const int* __restrict__ counts, const int* __restrict__ offsets) {
  int e = blockIdx.y >> 5;
  int t = blockIdx.y & (MAXT - 1);
  int cnt = counts[e];
  if (t * TM >= cnt) return;
  int row0 = offsets[e] + t * TM;
  int colblk = blockIdx.x * TN;

  __shared__ unsigned short Bs[2][TN * KPAD];

  int tid = threadIdx.x;
  int wave = tid >> 5, lane = tid & 31;
  int wrow = (wave >> 1) * 16;
  int wcol = (wave & 1) * 32;

  const float* w2e = w2 + (size_t)e * FFN * DIM + colblk;
  const unsigned short* arow = hbuf + (size_t)(row0 + wrow) * FFN;

  int rp = tid >> 4;
  int sc = (tid & 15) * 4;

  v8f acc[2] = {vzero(), vzero()};

  {
    size_t g0 = (size_t)(2 * rp) * DIM + sc;
    stage_pair(Bs[0], rp, sc, *reinterpret_cast<const float4*>(w2e + g0),
               *reinterpret_cast<const float4*>(w2e + g0 + DIM));
  }
  __syncthreads();

  for (int k0 = 0; k0 < FFN; k0 += KSTEP) {
    int cur = (k0 >> 5) & 1;
    bool has_next = (k0 + KSTEP) < FFN;
    float4 n2a, n2b;
    if (has_next) {
      size_t g0 = (size_t)(k0 + KSTEP + 2 * rp) * DIM + sc;
      n2a = *reinterpret_cast<const float4*>(w2e + g0);
      n2b = *reinterpret_cast<const float4*>(w2e + g0 + DIM);
      if (k0 + 2 * KSTEP < FFN)
        __builtin_prefetch(w2e + (size_t)(k0 + 2 * KSTEP) * DIM, 0, 1);
    }
    v16bf a  = load_frag_a(arow + k0, FFN, lane);
    v16bf ba = load_frag_b(&Bs[cur][(wcol + 0) * KPAD], lane);
    v16bf bb = load_frag_b(&Bs[cur][(wcol + 16) * KPAD], lane);
    acc[0] = wmma_bf16(a, ba, acc[0]);
    acc[1] = wmma_bf16(a, bb, acc[1]);
    if (has_next) stage_pair(Bs[cur ^ 1], rp, sc, n2a, n2b);
    __syncthreads();
  }
  // weighted scatter-combine; padded slots have score 0 / token 0 -> add 0
  int rbase = row0 + wrow + ((lane < 16) ? 0 : 8);
#pragma unroll
  for (int r = 0; r < 8; ++r) {
    int slot = rbase + r;
    float scw = slot_score[slot];
    int tok = slot_token[slot];
#pragma unroll
    for (int f = 0; f < 2; ++f) {
      int col = colblk + wcol + f * 16 + (lane & 15);
      atomicAdd(&out[(size_t)tok * DIM + col], acc[f][r] * scw);
    }
  }
}

extern "C" void kernel_launch(void* const* d_in, const int* in_sizes, int n_in,
                              void* d_out, int out_size, void* d_ws, size_t ws_size,
                              hipStream_t stream) {
  (void)in_sizes; (void)n_in; (void)out_size; (void)ws_size;
  const float* x      = (const float*)d_in[0];   // [1,1024,1024]
  const float* gate_w = (const float*)d_in[1];   // [8,1024]
  const float* w1     = (const float*)d_in[2];   // [8,1024,2048]
  const float* w2     = (const float*)d_in[3];   // [8,2048,1024]
  const float* w3     = (const float*)d_in[4];   // [8,1024,2048]
  float* out = (float*)d_out;                    // [1,1024,1024]

  // workspace layout (bytes)
  char* ws = (char*)d_ws;
  int*   counts     = (int*)(ws + 0);            //     8 i32
  int*   cursors    = (int*)(ws + 32);           //     8 i32
  int*   offsets    = (int*)(ws + 64);           //     8 i32
  int*   slot_token = (int*)(ws + 128);          //  2560 i32 -> ends 10368
  float* slot_score = (float*)(ws + 10368);      //  2560 f32 -> ends 20608
  unsigned short* x_sorted = (unsigned short*)(ws + 20608);  // 2560x1024 bf16
  int*   tok_sel    = (int*)(ws + 5263488);      // 1024x2 i32
  float* tok_score  = (float*)(ws + 5271680);    // 1024x2 f32
  unsigned short* hbuf = (unsigned short*)(ws + 5279872);    // 2560x2048 bf16
  // total ws use: 15,765,632 bytes

  // zero: counts/cursors/slot arrays/x_sorted padding, and out (scatter-add)
  zero_u32<<<2048, 256, 0, stream>>>((unsigned int*)ws, 5263488L / 4);
  zero_u32<<<2048, 256, 0, stream>>>((unsigned int*)out, (long)NTOK * DIM);

  moe_router<<<NTOK / 8, 256, 0, stream>>>(x, gate_w, tok_sel, tok_score, counts);
  moe_offsets<<<1, 32, 0, stream>>>(counts, offsets);
  moe_assign<<<(NTOK * TOPK) / 8, 256, 0, stream>>>(x, tok_sel, tok_score, offsets,
                                                    cursors, slot_token, slot_score,
                                                    x_sorted);
  moe_phase1<<<dim3(FFN / TN, E_EXPERTS * MAXT), 256, 0, stream>>>(
      x_sorted, w1, w3, hbuf, counts, offsets);
  moe_phase2<<<dim3(DIM / TN, E_EXPERTS * MAXT), 256, 0, stream>>>(
      hbuf, w2, slot_token, slot_score, out, counts, offsets);
}
